// ModernNCA_28527172780785
// MI455X (gfx1250) — compile-verified, matrix operand order checked by reading, not run on hardware
//
#include <hip/hip_runtime.h>
#include <hip/hip_bf16.h>

typedef __attribute__((ext_vector_type(8)))  _Float16 v8h;
typedef __attribute__((ext_vector_type(16))) _Float16 v16h;
typedef __attribute__((ext_vector_type(8)))  float    v8f;

// ---------------------------------------------------------------------------
// WMMA fragment loader (wave32, v_wmma_f32_16x16x32_f16 operand layout).
// Matrix is row-major f16 with leading dim `ld`. "row" selects the 16-row
// (or 16-col for the B operand, since B-frag(Wt) == A-frag(W row-major^T))
// tile base; k0 is the K-step base (multiple of 32).
// Lane l<16  : row=base+l,      halves k0+0..7  and k0+16..23
// Lane l>=16 : row=base+(l-16), halves k0+8..15 and k0+24..31
// Both chunks are 16-byte aligned -> global_load_b128.
// ---------------------------------------------------------------------------
__device__ inline v16h load_frag(const _Float16* __restrict__ base, int ld,
                                 int row0, int k0, int lane) {
    int r  = row0 + (lane & 15);
    int kA = k0 + ((lane & 16) ? 8 : 0);
    const _Float16* p = base + (size_t)r * (size_t)ld + kA;
    v8h lo = *(const v8h*)(p);
    v8h hi = *(const v8h*)(p + 16);
    v16h f;
#pragma unroll
    for (int i = 0; i < 8; ++i) { f[i] = lo[i]; f[i + 8] = hi[i]; }
    return f;
}

// ---------------------------------------------------------------------------
// Elementwise f32 -> f16 convert
// ---------------------------------------------------------------------------
__global__ void cvt_f16_kernel(const float* __restrict__ in,
                               _Float16* __restrict__ out, int n) {
    int i = blockIdx.x * blockDim.x + threadIdx.x;
    if (i < n) out[i] = (_Float16)in[i];
}

// ---------------------------------------------------------------------------
// Transpose + convert: in [K x N] f32 row-major -> out [N x K] f16 row-major
// ---------------------------------------------------------------------------
__global__ void transpose_cvt_kernel(const float* __restrict__ in,
                                     _Float16* __restrict__ outT,
                                     int K, int N) {
    int idx = blockIdx.x * blockDim.x + threadIdx.x;
    if (idx < K * N) {
        int k = idx / N;
        int n = idx - k * N;
        outT[n * K + k] = (_Float16)in[idx];
    }
}

// ---------------------------------------------------------------------------
// C16[M x N] = relu(A16[M x K] @ WT16[N x K]^T + bias)
// Block: 256 threads = 8 waves; wave w owns 16x16 tile at col blockIdx.y*128+w*16.
// Requires M % 16 == 0, N % 128 == 0 (on N>=128) , K % 32 == 0. All satisfied:
// M in {1024, 30000}, N in {512, 128}, K in {256, 512}.
// ---------------------------------------------------------------------------
__global__ void gemm_bias_relu_f16(const _Float16* __restrict__ A,
                                   const _Float16* __restrict__ WT,
                                   const float* __restrict__ bias,
                                   _Float16* __restrict__ C,
                                   int Nn, int K) {
    int lane = threadIdx.x & 31;
    int w    = threadIdx.x >> 5;
    int row0 = blockIdx.x * 16;
    int col0 = blockIdx.y * 128 + w * 16;

    v8f acc = {};
    for (int k0 = 0; k0 < K; k0 += 32) {
        v16h a = load_frag(A,  K, row0, k0, lane);
        v16h b = load_frag(WT, K, col0, k0, lane);
        acc = __builtin_amdgcn_wmma_f32_16x16x32_f16(
            false, a, false, b, (short)0, acc, false, false);
    }

    // C layout: VGPR r -> (M = r | r+8 for upper half-lanes, N = lane&15)
    int col   = col0 + (lane & 15);
    float bv  = bias[col];
    int rbase = row0 + ((lane & 16) ? 8 : 0);
#pragma unroll
    for (int r = 0; r < 8; ++r) {
        float v = acc[r] + bv;
        v = v > 0.0f ? v : 0.0f;
        C[(size_t)(rbase + r) * (size_t)Nn + col] = (_Float16)v;
    }
}

// ---------------------------------------------------------------------------
// out[m] = sum_j in16[m][j]^2   (Kd = 128)
// ---------------------------------------------------------------------------
__global__ void rowsumsq_f16(const _Float16* __restrict__ in,
                             float* __restrict__ out, int M, int Kd) {
    int m = blockIdx.x * blockDim.x + threadIdx.x;
    if (m < M) {
        const _Float16* p = in + (size_t)m * (size_t)Kd;
        float s = 0.0f;
        for (int j = 0; j < Kd; ++j) {
            float v = (float)p[j];
            s += v * v;
        }
        out[m] = s;
    }
}

// ---------------------------------------------------------------------------
// Fused NCA tail:
//   dot   = h @ sh^T (WMMA, K=128)
//   d2    = hsq + ssq - 2*dot ; dist = sqrt(max(d2,0)+eps)
//   wgt   = exp(-dist)                      (all logits <= 0: no max needed)
//   out[m]= sum(wgt*sub_y) / sum(wgt)
// Block handles 16 rows of h; 8 waves sweep N=30000 sub rows, 16 cols/tile.
// Requires B % 16 == 0, Nsub % 16 == 0, Kd == 128. (1024, 30000 ok)
// ---------------------------------------------------------------------------
__global__ void nca_softmax_kernel(const _Float16* __restrict__ h16,   // [B x 128]
                                   const _Float16* __restrict__ sh16,  // [N x 128]
                                   const float* __restrict__ hsq,      // [B]
                                   const float* __restrict__ ssq,      // [N]
                                   const float* __restrict__ suby,     // [N]
                                   float* __restrict__ out,            // [B]
                                   int Nsub) {
    const int KD = 128;
    int lane = threadIdx.x & 31;
    int w    = threadIdx.x >> 5;
    int row0 = blockIdx.x * 16;

    // Preload the 4 K-step A fragments of this block's 16 h-rows (constant
    // across the whole N sweep) and the per-lane hsq values.
    v16h afrag[4];
#pragma unroll
    for (int s = 0; s < 4; ++s) afrag[s] = load_frag(h16, KD, row0, s * 32, lane);

    int rbase = row0 + ((lane & 16) ? 8 : 0);
    float hq[8];
#pragma unroll
    for (int r = 0; r < 8; ++r) hq[r] = hsq[rbase + r];

    float num[8], den[8];
#pragma unroll
    for (int r = 0; r < 8; ++r) { num[r] = 0.0f; den[r] = 0.0f; }

    int ntiles = Nsub >> 4;             // 1875 column tiles of 16
    int iters  = (ntiles + 7) >> 3;     // per-wave iterations (wave-uniform)
    for (int it = 0; it < iters; ++it) {
        int t = it * 8 + w;             // wave-uniform -> EXEC stays full
        if (t >= ntiles) break;
        int c0 = t * 16;

        if (t + 8 < ntiles)             // pull next sh tile toward the WGP
            __builtin_prefetch(sh16 + (size_t)(c0 + 128) * KD, 0, 1);

        v8f acc = {};
#pragma unroll
        for (int s = 0; s < 4; ++s) {
            v16h b = load_frag(sh16, KD, c0, s * 32, lane);
            acc = __builtin_amdgcn_wmma_f32_16x16x32_f16(
                false, afrag[s], false, b, (short)0, acc, false, false);
        }

        int col  = c0 + (lane & 15);
        float sy = suby[col];
        float sq = ssq[col];
#pragma unroll
        for (int r = 0; r < 8; ++r) {
            float d2 = hq[r] + sq - 2.0f * acc[r];
            d2 = d2 > 0.0f ? d2 : 0.0f;
            float dist = __builtin_sqrtf(d2 + 1e-8f);
            float wgt  = __expf(-dist);
            num[r] += wgt * sy;
            den[r] += wgt;
        }
    }

    // Reduce the per-lane partials (each lane holds 8 distinct rows) in LDS.
    __shared__ float snum[16];
    __shared__ float sden[16];
    if (threadIdx.x < 16) { snum[threadIdx.x] = 0.0f; sden[threadIdx.x] = 0.0f; }
    __syncthreads();

    int rl = (lane & 16) ? 8 : 0;
#pragma unroll
    for (int r = 0; r < 8; ++r) {
        atomicAdd(&snum[rl + r], num[r]);   // ds_add_f32
        atomicAdd(&sden[rl + r], den[r]);
    }
    __syncthreads();

    if (threadIdx.x < 16)
        out[row0 + threadIdx.x] = snum[threadIdx.x] / sden[threadIdx.x];
}

// ---------------------------------------------------------------------------
// Host-side orchestration
// ---------------------------------------------------------------------------
extern "C" void kernel_launch(void* const* d_in, const int* in_sizes, int n_in,
                              void* d_out, int out_size, void* d_ws, size_t ws_size,
                              hipStream_t stream) {
    const float* x     = (const float*)d_in[0];
    const float* sub_x = (const float*)d_in[1];
    const float* sub_y = (const float*)d_in[2];
    const float* W1    = (const float*)d_in[3];
    const float* b1    = (const float*)d_in[4];
    const float* W2    = (const float*)d_in[5];
    const float* b2    = (const float*)d_in[6];

    const int H1 = in_sizes[4];            // 512
    const int H2 = in_sizes[6];            // 128
    const int D  = in_sizes[3] / H1;       // 256
    const int B  = in_sizes[0] / D;        // 1024
    const int N  = in_sizes[2];            // 30000

    char* ws = (char*)d_ws;
    size_t off = 0;
    auto alloc = [&](size_t bytes) -> void* {
        void* p = ws + off;
        off += (bytes + 255) & ~(size_t)255;
        return p;
    };

    _Float16* x16  = (_Float16*)alloc((size_t)B * D * 2);
    _Float16* sx16 = (_Float16*)alloc((size_t)N * D * 2);
    _Float16* w1t  = (_Float16*)alloc((size_t)H1 * D * 2);
    _Float16* w2t  = (_Float16*)alloc((size_t)H2 * H1 * 2);
    _Float16* h1x  = (_Float16*)alloc((size_t)B * H1 * 2);
    _Float16* h1s  = (_Float16*)alloc((size_t)N * H1 * 2);
    _Float16* hx   = (_Float16*)alloc((size_t)B * H2 * 2);
    _Float16* hs   = (_Float16*)alloc((size_t)N * H2 * 2);
    float*    hsq  = (float*)alloc((size_t)B * 4);
    float*    ssq  = (float*)alloc((size_t)N * 4);

    const int T = 256;
    auto cdiv = [](int a, int b) { return (a + b - 1) / b; };

    // 1) f16 conversions of activations and transposed weights
    cvt_f16_kernel<<<cdiv(B * D, T), T, 0, stream>>>(x, x16, B * D);
    cvt_f16_kernel<<<cdiv(N * D, T), T, 0, stream>>>(sub_x, sx16, N * D);
    transpose_cvt_kernel<<<cdiv(D * H1, T), T, 0, stream>>>(W1, w1t, D, H1);
    transpose_cvt_kernel<<<cdiv(H1 * H2, T), T, 0, stream>>>(W2, w2t, H1, H2);

    // 2) MLP layer 1: relu(X @ W1 + b1) -> [*, H1]
    gemm_bias_relu_f16<<<dim3(B / 16, H1 / 128), T, 0, stream>>>(x16,  w1t, b1, h1x, H1, D);
    gemm_bias_relu_f16<<<dim3(N / 16, H1 / 128), T, 0, stream>>>(sx16, w1t, b1, h1s, H1, D);

    // 3) MLP layer 2: relu(H1 @ W2 + b2) -> [*, H2]
    gemm_bias_relu_f16<<<dim3(B / 16, H2 / 128), T, 0, stream>>>(h1x, w2t, b2, hx, H2, H1);
    gemm_bias_relu_f16<<<dim3(N / 16, H2 / 128), T, 0, stream>>>(h1s, w2t, b2, hs, H2, H1);

    // 4) squared norms of the f16-rounded features (consistent with the GEMM)
    rowsumsq_f16<<<cdiv(B, T), T, 0, stream>>>(hx, hsq, B, H2);
    rowsumsq_f16<<<cdiv(N, T), T, 0, stream>>>(hs, ssq, N, H2);

    // 5) fused distance / exp / weighted-average
    nca_softmax_kernel<<<B / 16, T, 0, stream>>>(hx, hs, hsq, ssq, sub_y,
                                                 (float*)d_out, N);
}
